// CFConv_4741643894827
// MI455X (gfx1250) — matrix operand order repaired and checked by reading, not compiled, not run on hardware
//
#include <hip/hip_runtime.h>

// CFConv (SchNet continuous-filter conv) for gfx1250 / MI455X.
// Per (b,i): rbf(192x64) @ W1(64x128) -> ssp -> @ W2(128x128) -> ssp,
// then out[b,i,g] = sum_j H2[j,g] * X[b,j,g], masked.
// GEMMs on v_wmma_f32_16x16x32_f16 (f16 operands, f32 accumulate).
// Weights pre-converted to f16 + transposed ONCE into d_ws by a prep kernel.
// Each wave keeps its loop-invariant WMMA B-fragments in registers (48 VGPRs),
// loaded once from the workspace with contiguous b128 global loads -> only the
// A-fragments (rbf / H1 tiles) flow through LDS in the main loop.

typedef _Float16 h16 __attribute__((ext_vector_type(16)));
typedef _Float16 h8  __attribute__((ext_vector_type(8)));
typedef float    vf8 __attribute__((ext_vector_type(8)));

union H16 { h16 v; h8 h[2]; };

#define CF_BS   8
#define CF_N    192
#define CF_FD   128
#define CF_NB   64
#define CF_GAMMA 10.0f
#define CF_LN2  0.6931471805599453f
#define CF_MUSTEP (30.0f / 63.0f)       // linspace(0,30,64) step
#define CF_W1T_ELTS (CF_FD * CF_NB)     // 8192  f16, layout [f][k]
#define CF_W2T_ELTS (CF_FD * CF_FD)     // 16384 f16, layout [g][f]
#define CF_WTS_ELTS (CF_W1T_ELTS + CF_W2T_ELTS)
#define CF_WTS_BYTES (CF_WTS_ELTS * 2)  // 49152 B

// shifted softplus: log(1+exp(x)) - log(2), numerically stable.
// Native v_exp_f32 / v_log_f32; abs error ~1e-7, negligible vs f16 operands.
__device__ __forceinline__ float cf_ssp(float x) {
    return fmaxf(x, 0.0f) + __logf(1.0f + __expf(-fabsf(x))) - CF_LN2;
}

// ---- prep: convert + transpose weights to f16 once (64 blocks x 256) ----
__global__ __launch_bounds__(256)
void cf_prep_kernel(const float* __restrict__ W1,
                    const float* __restrict__ W2,
                    _Float16* __restrict__ wts) {
    int idx = blockIdx.x * 256 + threadIdx.x;          // 0 .. 16383
    if (idx < CF_NB * CF_FD) {                         // W1[k][f] -> wts[f*64+k]
        int k = idx / CF_FD, f = idx % CF_FD;
        wts[f * CF_NB + k] = (_Float16)W1[idx];
    }
    if (idx < CF_FD * CF_FD) {                         // W2[f][g] -> wts[8192 + g*128+f]
        int f = idx / CF_FD, g = idx % CF_FD;
        wts[CF_W1T_ELTS + g * CF_FD + f] = (_Float16)W2[idx];
    }
}

__global__ __launch_bounds__(256)
void cfconv_wmma_kernel(const float* __restrict__ X,
                        const float* __restrict__ R,
                        const float* __restrict__ Mask,
                        const float* __restrict__ W1,
                        const float* __restrict__ b1,
                        const float* __restrict__ W2,
                        const float* __restrict__ b2,
                        const _Float16* __restrict__ wts,
                        int use_ws,
                        float* __restrict__ out) {
    // One block per (b, i). 256 threads = 8 wave32s.
    const int bi   = blockIdx.x;
    const int b    = bi / CF_N;
    const int i    = bi % CF_N;
    const int t    = threadIdx.x;
    const int lane = t & 31;
    const int wave = t >> 5;          // 0..7: owns f-tile (GEMM1) and g-tile (GEMM2)
    const int ln16 = lane & 15;

    // WMMA per-lane layout as used here (CDNA5 wave32; A/B K order contiguous, not the ISA table):
    //  A 16x32 f16: row M = lane%16; halves e   -> K = hs16+e (same order as B)
    //  B 32x16 f16: col N = lane%16; halves e   -> K = hs16+e
    //  C/D 16x16 f32: col N = lane%16; VGPR r -> row M = r + hs8
    const int hs8  = (lane < 16) ? 0 : 8;
    const int hs16 = (lane < 16) ? 0 : 16;
    const int col  = wave * 16 + ln16;   // fixed output column for this thread

    // Small LDS footprint: only the per-tile A operands + distances.
    __shared__ __align__(16) _Float16 sH1 [16 * CF_FD];   // [jr][f]  4 KB
    __shared__ __align__(16) _Float16 sRbf[16 * CF_NB];   // [jr][k]  2 KB
    __shared__ float sDist[CF_N];                         //         768 B

    // ---- distances (per-block, depends on i) ----
    if (t < CF_N) {
        const float* Ri = R + (size_t)(b * CF_N + i) * 3;
        const float* Rj = R + (size_t)(b * CF_N + t) * 3;
        float dx = Ri[0] - Rj[0], dy = Ri[1] - Rj[1], dz = Ri[2] - Rj[2];
        float d2 = dx * dx + dy * dy + dz * dz;
        sDist[t] = (d2 > 0.0f) ? sqrtf(d2) : 0.0f;
    }

    // ---- loop-invariant WMMA B-fragments, held in registers ----
    H16 bw1[2];   // W1t slices: K = kt*32 + hs16 + e,  col fixed
    H16 bw2[4];   // W2t slices
    if (use_ws) {
        const _Float16* gW1t = wts;                 // [f][k], k contiguous
        const _Float16* gW2t = wts + CF_W1T_ELTS;   // [g][f], f contiguous
        #pragma unroll
        for (int kt = 0; kt < 2; ++kt) {
            const _Float16* p = &gW1t[col * CF_NB + kt * 32 + hs16]; // 32B aligned
            bw1[kt].h[0] = *(const h8*)p;
            bw1[kt].h[1] = *(const h8*)(p + 8);
        }
        #pragma unroll
        for (int kt = 0; kt < 4; ++kt) {
            const _Float16* p = &gW2t[col * CF_FD + kt * 32 + hs16];
            bw2[kt].h[0] = *(const h8*)p;
            bw2[kt].h[1] = *(const h8*)(p + 8);
        }
    } else {
        // fallback: strided gather + convert from f32 weights
        #pragma unroll
        for (int kt = 0; kt < 2; ++kt)
            #pragma unroll
            for (int e = 0; e < 16; ++e) {
                int k = kt * 32 + hs16 + e;
                bw1[kt].v[e] = (_Float16)W1[k * CF_FD + col];
            }
        #pragma unroll
        for (int kt = 0; kt < 4; ++kt)
            #pragma unroll
            for (int e = 0; e < 16; ++e) {
                int f = kt * 32 + hs16 + e;
                bw2[kt].v[e] = (_Float16)W2[f * CF_FD + col];
            }
    }
    __syncthreads();

    const float b1v = b1[col];
    const float b2v = b2[col];
    const float* Xb = X + (size_t)b * CF_N * CF_FD;

    float outp = 0.0f;  // partial of out[b,i,col] over j rows this thread touches

    for (int jt = 0; jt < CF_N / 16; ++jt) {
        // cover the epilogue's X loads for this tile (global_prefetch_b8)
        __builtin_prefetch(&Xb[(jt * 16 + hs8) * CF_FD + col], 0, 3);

        // ---- rbf tile 16x64 (computed once, shared by all 8 waves) ----
        #pragma unroll
        for (int q = 0; q < 4; ++q) {
            int lin = t * 4 + q;                 // 0..1023
            int jr = lin >> 6, k = lin & 63;
            float u = (float)k * CF_MUSTEP - sDist[jt * 16 + jr];
            sRbf[lin] = (_Float16)__expf(-CF_GAMMA * u * u);
        }
        __syncthreads();

        // ---- GEMM1: (16x64 rbf) x (64x16 W1 slice), wave owns f-tile = wave ----
        vf8 c1 = {};
        #pragma unroll
        for (int kt = 0; kt < 2; ++kt) {
            H16 a;
            const int ab = ln16 * CF_NB + kt * 32 + hs16;
            a.h[0] = *(const h8*)&sRbf[ab];
            a.h[1] = *(const h8*)&sRbf[ab + 8];
            c1 = __builtin_amdgcn_wmma_f32_16x16x32_f16(
                     false, a.v, false, bw1[kt].v, (short)0, c1, false, false);
        }
        // bias + shifted-softplus, stage H1 tile as f16 (A operand for GEMM2)
        #pragma unroll
        for (int r = 0; r < 8; ++r) {
            int row = r + hs8;
            sH1[row * CF_FD + col] = (_Float16)cf_ssp(c1[r] + b1v);
        }
        __syncthreads();

        // ---- GEMM2: (16x128 H1) x (128x16 W2 slice), wave owns g-tile = wave ----
        vf8 c2 = {};
        #pragma unroll
        for (int kt = 0; kt < 4; ++kt) {
            H16 a;
            const int ab = ln16 * CF_FD + kt * 32 + hs16;
            a.h[0] = *(const h8*)&sH1[ab];
            a.h[1] = *(const h8*)&sH1[ab + 8];
            c2 = __builtin_amdgcn_wmma_f32_16x16x32_f16(
                     false, a.v, false, bw2[kt].v, (short)0, c2, false, false);
        }
        // bias + ssp, weight by X[b, j, col], reduce over j into register
        #pragma unroll
        for (int r = 0; r < 8; ++r) {
            int j = jt * 16 + r + hs8;
            float w = cf_ssp(c2[r] + b2v);
            outp += w * Xb[j * CF_FD + col];
        }
        __syncthreads();   // protect sRbf / sH1 before next iteration overwrites
    }

    // Lanes L and L+16 hold the same column; columns are distinct across waves.
    outp += __shfl_down(outp, 16);
    if (lane < 16) {
        out[(size_t)(b * CF_N + i) * CF_FD + col] = outp * Mask[b * CF_N + i];
    }
}

extern "C" void kernel_launch(void* const* d_in, const int* in_sizes, int n_in,
                              void* d_out, int out_size, void* d_ws, size_t ws_size,
                              hipStream_t stream) {
    (void)in_sizes; (void)n_in; (void)out_size;
    const float* X    = (const float*)d_in[0];
    const float* R    = (const float*)d_in[1];
    const float* Mask = (const float*)d_in[2];
    const float* W1   = (const float*)d_in[3];
    const float* b1   = (const float*)d_in[4];
    const float* W2   = (const float*)d_in[5];
    const float* b2   = (const float*)d_in[6];
    float* out = (float*)d_out;

    const int use_ws = (d_ws != nullptr && ws_size >= CF_WTS_BYTES) ? 1 : 0;
    _Float16* wts = (_Float16*)d_ws;

    if (use_ws) {
        // one-time f32 -> f16 transpose of W1/W2 into workspace
        cf_prep_kernel<<<dim3(64), dim3(256), 0, stream>>>(W1, W2, wts);
    }

    dim3 grid(CF_BS * CF_N);   // 1536 workgroups: one per (b, i)
    dim3 block(256);           // 8 wave32s
    cfconv_wmma_kernel<<<grid, block, 0, stream>>>(X, R, Mask, W1, b1, W2, b2,
                                                   wts, use_ws, out);
}
